// AdditiveAttention_89206470738083
// MI455X (gfx1250) — matrix operand order repaired and checked by reading, not compiled, not run on hardware
//
#include <hip/hip_runtime.h>
#include <hip/hip_bf16.h>

// Problem constants (match reference)
#define B_   4
#define Q_   256
#define K_   1024
#define D_   512     // QD == KD
#define H_   256
#define DV_  512
#define NEG_INF_ (-1e9f)

typedef __attribute__((ext_vector_type(16))) __bf16 v16bf;
typedef __attribute__((ext_vector_type(8)))  float  v8f;

// ---- CDNA5 feature probes (compile-time, with safe fallbacks) -------------
#if defined(__has_builtin)
#  if __has_builtin(__builtin_amdgcn_global_load_async_to_lds_b32) && \
      __has_builtin(__builtin_amdgcn_s_wait_asynccnt)
#    define USE_ASYNC_LDS 1
#  endif
#  if __has_builtin(__builtin_amdgcn_tanhf)
#    define TANH_FAST(x) __builtin_amdgcn_tanhf(x)
#  elif __has_builtin(__builtin_amdgcn_tanh_f32)
#    define TANH_FAST(x) __builtin_amdgcn_tanh_f32(x)
#  endif
#endif
#if !defined(TANH_FAST)
#  if defined(__AMDGCN__)
#    warning "CDNA5: no v_tanh builtin found -- using exp-based tanh fallback"
#  endif
#  define TANH_FAST(x) (1.f - 2.f / (__expf(2.f * (x)) + 1.f))
#endif
#if !defined(USE_ASYNC_LDS) && defined(__AMDGCN__)
#  warning "CDNA5: async-to-LDS builtins unavailable -- using direct LDS staging"
#endif

// ---------------------------------------------------------------------------
// Kernel 1: C[M,N] = X[M,Dk] * W[N,Dk]^T  via v_wmma_f32_16x16x32_bf16.
// One wave computes one 16x16 tile; K-loop steps of 32, unrolled x2.
// A/B fragments use contiguous k = 16*half + s in both operands (wave32).
// ---------------------------------------------------------------------------
__global__ __launch_bounds__(128) void proj_gemm_bf16_wmma(
    const float* __restrict__ X, const float* __restrict__ W,
    float* __restrict__ C, int M, int Dk, int N) {
  const int wid    = (blockIdx.x * blockDim.x + threadIdx.x) >> 5;
  const int lane   = threadIdx.x & 31;
  const int tilesN = N >> 4;
  const int tileM  = wid / tilesN;
  const int tileN  = wid - tileM * tilesN;
  if (tileM * 16 >= M) return;                 // wave-uniform guard
  const int m0 = tileM << 4, n0 = tileN << 4;
  const int half = lane >> 4;                  // 0 or 1
  const int l15  = lane & 15;

  const float* xrow = X + (size_t)(m0 + l15) * Dk;
  const float* wrow = W + (size_t)(n0 + l15) * Dk;

  v8f c = {};
#pragma unroll 2
  for (int kk = 0; kk < Dk; kk += 32) {
    v16bf a, b;
    const int aoff0 = kk + (half ? 16 : 0);
    const int aoff1 = kk + (half ? 24 : 8);
#pragma unroll
    for (int i = 0; i < 8; ++i) {
      a[i]     = (__bf16)xrow[aoff0 + i];
      a[i + 8] = (__bf16)xrow[aoff1 + i];
    }
    const int boff = kk + (half ? 16 : 0);
#pragma unroll
    for (int i = 0; i < 16; ++i) b[i] = (__bf16)wrow[boff + i];
    c = __builtin_amdgcn_wmma_f32_16x16x32_bf16(false, a, false, b,
                                                (short)0, c, false, false);
  }
  const int rbase = m0 + (half ? 8 : 0);
#pragma unroll
  for (int j = 0; j < 8; ++j)
    C[(size_t)(rbase + j) * N + n0 + l15] = c[j];
}

// ---------------------------------------------------------------------------
// Kernel 2: per (b, q-block of 4): u[q][k] = sum_h Vw[h]*tanh(qp[q][h]+kp[k][h])
// then masked softmax over k. Double-buffered async-to-LDS staging of kp
// chunks (16 rows x 256 h), 4 q-rows amortize every staged byte.
// ---------------------------------------------------------------------------
#define CH_   16                   // kp rows per chunk
#define NCH_  (K_ / CH_)           // 64 chunks
#define QB_   4                    // q rows per block
#define KPAD_ 257                  // row stride in LDS (bank-conflict-free)

__device__ __forceinline__ void stage_kp(const float* __restrict__ gsrc,
                                         float* lbuf, int tid) {
#if defined(USE_ASYNC_LDS) && defined(__AMDGCN__)
#pragma unroll
  for (int t = 0; t < CH_; ++t) {
    __builtin_amdgcn_global_load_async_to_lds_b32(
        (__attribute__((address_space(1))) int*)(void*)(gsrc + (size_t)t * H_ + tid),
        (__attribute__((address_space(3))) int*)(lbuf + t * KPAD_ + tid),
        0, 0);
  }
#else
#pragma unroll 4
  for (int t = 0; t < CH_; ++t)
    lbuf[t * KPAD_ + tid] = gsrc[(size_t)t * H_ + tid];
#endif
}

__device__ __forceinline__ void wait_stage() {
#if defined(USE_ASYNC_LDS) && defined(__AMDGCN__)
  __builtin_amdgcn_s_wait_asynccnt(0);
#endif
}

__global__ __launch_bounds__(256) void additive_scores_softmax(
    const float* __restrict__ qp, const float* __restrict__ kp,
    const float* __restrict__ Vw, const int* __restrict__ rep_mask,
    float* __restrict__ attn_out) {
  __shared__ float s_kp[2][CH_ * KPAD_];     // 2 x 16 x 257 floats = 32.9KB
  __shared__ float s_qp[QB_][H_];            // 4KB
  __shared__ float s_vw[H_];                 // 1KB
  __shared__ float s_u[QB_][K_];             // 16KB
  __shared__ float s_part[QB_ * 16 * 16];    // 4KB
  __shared__ float s_red[256];               // 1KB

  const int tid = threadIdx.x;
  const int b   = blockIdx.x / (Q_ / QB_);
  const int q0  = (blockIdx.x - b * (Q_ / QB_)) * QB_;

#pragma unroll
  for (int r = 0; r < QB_; ++r)
    s_qp[r][tid] = qp[(size_t)(b * Q_ + q0 + r) * H_ + tid];
  s_vw[tid] = Vw[tid];

  const float* kpb = kp + (size_t)b * K_ * H_;
  stage_kp(kpb, s_kp[0], tid);               // prefetch chunk 0

  const int part = tid >> 4;                 // 0..15 -> h block of 16
  const int kloc = tid & 15;                 // 0..15 -> k row in chunk
  const int h0   = part << 4;

  for (int c = 0; c < NCH_; ++c) {
    const int cur = c & 1;
    wait_stage();                            // this wave's async copies done
    __syncthreads();                         // all waves' copies visible
    if (c + 1 < NCH_)
      stage_kp(kpb + (size_t)(c + 1) * CH_ * H_, s_kp[cur ^ 1], tid);

    const float* krow = &s_kp[cur][kloc * KPAD_];
    float a0 = 0.f, a1 = 0.f, a2 = 0.f, a3 = 0.f;
#pragma unroll 4
    for (int h = h0; h < h0 + 16; ++h) {
      const float kv = krow[h];
      const float w  = s_vw[h];
      a0 += w * TANH_FAST(s_qp[0][h] + kv);
      a1 += w * TANH_FAST(s_qp[1][h] + kv);
      a2 += w * TANH_FAST(s_qp[2][h] + kv);
      a3 += w * TANH_FAST(s_qp[3][h] + kv);
    }
    s_part[(0 * 16 + part) * 16 + kloc] = a0;
    s_part[(1 * 16 + part) * 16 + kloc] = a1;
    s_part[(2 * 16 + part) * 16 + kloc] = a2;
    s_part[(3 * 16 + part) * 16 + kloc] = a3;
    __syncthreads();
    if (tid < QB_ * CH_) {                   // 64 threads finalize
      const int qq = tid >> 4;
      const int k  = tid & 15;
      float u = 0.f;
#pragma unroll
      for (int p = 0; p < 16; ++p) u += s_part[(qq * 16 + p) * 16 + k];
      const int kg = c * CH_ + k;
      s_u[qq][kg] = (rep_mask[b * K_ + kg] != 0) ? u : NEG_INF_;
    }
    // next iteration's top-of-loop barrier closes this chunk
  }
  __syncthreads();

  // ---- masked softmax per q row (4 elems per thread per row) ----
  for (int r = 0; r < QB_; ++r) {
    float lmax = -3.4e38f;
#pragma unroll
    for (int j = 0; j < 4; ++j) lmax = fmaxf(lmax, s_u[r][tid + j * 256]);
    s_red[tid] = lmax;
    __syncthreads();
    for (int s = 128; s > 0; s >>= 1) {
      if (tid < s) s_red[tid] = fmaxf(s_red[tid], s_red[tid + s]);
      __syncthreads();
    }
    const float rmax = s_red[0];
    __syncthreads();

    float e[4];
    float lsum = 0.f;
#pragma unroll
    for (int j = 0; j < 4; ++j) {
      e[j] = __expf(s_u[r][tid + j * 256] - rmax);
      lsum += e[j];
    }
    s_red[tid] = lsum;
    __syncthreads();
    for (int s = 128; s > 0; s >>= 1) {
      if (tid < s) s_red[tid] += s_red[tid + s];
      __syncthreads();
    }
    const float inv = 1.f / s_red[0];

    float* arow = attn_out + (size_t)(b * Q_ + q0 + r) * K_;
#pragma unroll
    for (int j = 0; j < 4; ++j) arow[tid + j * 256] = e[j] * inv;
    __syncthreads();                         // s_red reused next row
  }
}

// ---------------------------------------------------------------------------
// Kernel 3: out[b] (Q x DV) = attn[b] (Q x K) * V[b] (K x DV), bf16 WMMA.
// B-fragment loads are lane-coalesced (16 consecutive floats per k-row).
// ---------------------------------------------------------------------------
__global__ __launch_bounds__(128) void av_gemm_bf16_wmma(
    const float* __restrict__ attn, const float* __restrict__ V,
    float* __restrict__ out) {
  const int wid      = (blockIdx.x * blockDim.x + threadIdx.x) >> 5;
  const int lane     = threadIdx.x & 31;
  const int tilesN   = DV_ / 16;            // 32
  const int tilesPer = (Q_ / 16) * tilesN;  // 512 per batch
  const int b     = wid / tilesPer;
  const int t     = wid - b * tilesPer;
  const int tileM = t / tilesN;
  const int tileN = t - tileM * tilesN;
  const int m0 = tileM << 4, n0 = tileN << 4;
  const int half = lane >> 4;
  const int l15  = lane & 15;

  const float* arow  = attn + (size_t)b * Q_ * K_ + (size_t)(m0 + l15) * K_;
  const float* vbase = V + (size_t)b * K_ * DV_;

  v8f c = {};
#pragma unroll 2
  for (int kk = 0; kk < K_; kk += 32) {
    v16bf a, bm;
    const int aoff0 = kk + (half ? 16 : 0);
    const int aoff1 = kk + (half ? 24 : 8);
#pragma unroll
    for (int i = 0; i < 8; ++i) {
      a[i]     = (__bf16)arow[aoff0 + i];
      a[i + 8] = (__bf16)arow[aoff1 + i];
    }
    const int boff = kk + (half ? 16 : 0);
#pragma unroll
    for (int i = 0; i < 16; ++i)
      bm[i] = (__bf16)vbase[(size_t)(boff + i) * DV_ + n0 + l15];
    c = __builtin_amdgcn_wmma_f32_16x16x32_bf16(false, a, false, bm,
                                                (short)0, c, false, false);
  }
  const int rbase = m0 + (half ? 8 : 0);
  float* obase = out + (size_t)b * Q_ * DV_;
#pragma unroll
  for (int j = 0; j < 8; ++j)
    obase[(size_t)(rbase + j) * DV_ + n0 + l15] = c[j];
}

// ---------------------------------------------------------------------------
extern "C" void kernel_launch(void* const* d_in, const int* in_sizes, int n_in,
                              void* d_out, int out_size, void* d_ws, size_t ws_size,
                              hipStream_t stream) {
  const float* q  = (const float*)d_in[0];
  const float* k  = (const float*)d_in[1];
  const float* v  = (const float*)d_in[2];
  const int*   rm = (const int*)d_in[3];
  const float* Wq = (const float*)d_in[4];
  const float* Wk = (const float*)d_in[5];
  const float* Vw = (const float*)d_in[6];
  (void)in_sizes; (void)n_in; (void)out_size; (void)ws_size;

  float* out_f = (float*)d_out;
  float* attn  = out_f + (size_t)B_ * Q_ * DV_;   // outputs concatenated flat

  float* qp = (float*)d_ws;                       // (B*Q, H) f32
  float* kp = qp + (size_t)B_ * Q_ * H_;          // (B*K, H) f32

  // q_proj: 1024 tile-waves, 4 waves/block
  {
    const int tiles = (B_ * Q_ / 16) * (H_ / 16);
    proj_gemm_bf16_wmma<<<tiles / 4, 128, 0, stream>>>(q, Wq, qp, B_ * Q_, D_, H_);
  }
  // k_proj: 4096 tile-waves
  {
    const int tiles = (B_ * K_ / 16) * (H_ / 16);
    proj_gemm_bf16_wmma<<<tiles / 4, 128, 0, stream>>>(k, Wk, kp, B_ * K_, D_, H_);
  }
  // scores + masked softmax -> attn (written directly into d_out)
  additive_scores_softmax<<<B_ * (Q_ / QB_), 256, 0, stream>>>(qp, kp, Vw, rm, attn);
  // out = attn @ v
  {
    const int waves = B_ * (Q_ / 16) * (DV_ / 16); // 2048
    av_gemm_bf16_wmma<<<waves / 4, 128, 0, stream>>>(attn, v, out_f);
  }
}
